// VectorQuantizer_18279380812227
// MI455X (gfx1250) — matrix operand order repaired and checked by reading, not compiled, not run on hardware
//
#include <hip/hip_runtime.h>
#include <hip/hip_bf16.h>

// ---------------------------------------------------------------------------
// VQ-VAE vector quantizer forward for MI455X (gfx1250, wave32, WMMA).
//   z      : [64, 256, 32, 32] f32   (B=64, C=256, H=W=32) -> N = 65536 rows
//   emb_w  : [1024, 256] f32         (K = 1024 codes)
//   out    : z_q [64,256,32,32] f32  (16777216 floats) + loss (1 float)
// Distance GEMM via v_wmma_f32_16x16x32_bf16 with bf16 hi/lo split
// (hi*hi + hi*lo + lo*hi -> fp32-class argmin accuracy), B tiles staged to
// LDS with async global->LDS copies (ASYNCcnt) + double buffering.
// ---------------------------------------------------------------------------

typedef __attribute__((ext_vector_type(16))) __bf16 v16bf;
typedef __attribute__((ext_vector_type(8)))  float  v8f;
typedef __attribute__((ext_vector_type(4)))  int    v4i;

#define N_EMB      1024
#define LATENT     256
#define BATCH      64
#define HWSZ       1024              // 32*32
#define NROWS      65536             // BATCH*HWSZ
#define KSTEPS     8                 // 256 / 32
#define NTILES     64                // 1024 / 16
#define WAVES_PER_BLOCK 8

// workspace layout (bytes)
#define WS_EHI     0u                // 262144 bf16 = 524288 B (B-fragment order)
#define WS_ELO     524288u           // 262144 bf16
#define WS_ESQ     1048576u         // 1024 f32
#define WS_IDX     1052672u         // 65536 u32
#define WS_PART    1314816u         // 256 f32

#if defined(__has_builtin)
#  if __has_builtin(__builtin_amdgcn_global_load_async_to_lds_b128)
#    define HAVE_ASYNC_LDS 1
#  endif
#endif
#ifndef HAVE_ASYNC_LDS
#  define HAVE_ASYNC_LDS 0
#endif

typedef __attribute__((address_space(1))) void* as1_ptr;
typedef __attribute__((address_space(1))) v4i*  as1_v4i;
typedef __attribute__((address_space(3))) void* as3_ptr;
typedef __attribute__((address_space(3))) v4i*  as3_v4i;

__device__ __forceinline__ void async_wait0() {
#if HAVE_ASYNC_LDS
#  if __has_builtin(__builtin_amdgcn_s_wait_asynccnt)
    __builtin_amdgcn_s_wait_asynccnt(0);
#  else
    asm volatile("s_wait_asynccnt 0x0" ::: "memory");
#  endif
#endif
}

__device__ __forceinline__ unsigned int ord_f32(float f) {
    unsigned int u = __float_as_uint(f);
    return (u & 0x80000000u) ? ~u : (u | 0x80000000u);
}

__device__ __forceinline__ unsigned long long shfl_xor_u64(unsigned long long v, int mask) {
    unsigned int lo = (unsigned int)v;
    unsigned int hi = (unsigned int)(v >> 32);
    lo = __shfl_xor(lo, mask, 32);
    hi = __shfl_xor(hi, mask, 32);
    return ((unsigned long long)hi << 32) | (unsigned long long)lo;
}

// ---------------------------------------------------------------------------
// Kernel 1: codebook prep. One block per code (1024 blocks x 256 threads).
//   - split emb_w into bf16 hi/lo, stored in WMMA B-fragment order:
//       ehi[((nt*8+ks)*32 + 16*((lane>>3)&1) + ncol)*16 + s] = bf16(emb[(nt*16+ncol)*256 + ks*32+lane])
//     (s = (lane&7) | ((lane>>4)<<3); fragment lane = 16*half + column, k in AMD table order)
//   - e_sq[code] = sum_d emb[code][d]^2  (deterministic tree reduce)
// ---------------------------------------------------------------------------
__global__ void vq_prep(const float* __restrict__ emb,
                        __bf16* __restrict__ ehi, __bf16* __restrict__ elo,
                        float* __restrict__ esq) {
    __shared__ float sm[256];
    const int code = blockIdx.x;     // 0..1023
    const int d    = threadIdx.x;    // 0..255
    float f = emb[code * LATENT + d];
    __bf16 h = (__bf16)f;
    float  lof = f - (float)h;

    const int nt   = code >> 4;
    const int ncol = code & 15;
    const int ks   = d >> 5;
    const int lane = d & 31;
    const unsigned off = ((unsigned)(nt * KSTEPS + ks) * 32u + (unsigned)(((lane >> 3) & 1) * 16 + ncol)) * 16u + (unsigned)((lane & 7) | ((lane >> 4) << 3));
    ehi[off] = h;
    elo[off] = (__bf16)lof;

    sm[d] = f * f;
    __syncthreads();
    for (int s = 128; s > 0; s >>= 1) {
        if (d < s) sm[d] += sm[d + s];
        __syncthreads();
    }
    if (d == 0) esq[code] = sm[0];
}

// ---------------------------------------------------------------------------
// Kernel 2: distances + argmin. 512 blocks x 256 threads (8 waves).
// Each wave owns a 16-row tile of z, holds A fragments (hi/lo bf16, 16x32
// layout) in registers, loops over 64 code tiles double-buffered in LDS via
// async global->LDS copies; 3 independent WMMA accumulator chains.
// ---------------------------------------------------------------------------
__global__ void __launch_bounds__(256)
vq_dist(const float* __restrict__ z,
        const __bf16* __restrict__ ehi, const __bf16* __restrict__ elo,
        const float* __restrict__ esq,
        unsigned int* __restrict__ idxout) {
    __shared__ uint4 lds4[2048];     // 32 KB: two 16 KB buffers (hi 8KB + lo 8KB each)

    const int tid  = threadIdx.x;
    const int lane = tid & 31;
    const int wave = tid >> 5;
    const int tile = blockIdx.x * WAVES_PER_BLOCK + wave;   // 0..4095
    const int b    = tile >> 6;                             // batch
    const int hw0  = (tile & 63) << 4;                      // first hw of tile
    const float* zb = z + (size_t)b * (LATENT * HWSZ) + hw0;

    const int m     = lane & 15;     // A-matrix row within tile
    const int khalf = lane >> 4;     // which K half-group this lane holds

    // ---- Build A fragments in registers (read z once, coalesced 64B groups).
    // 16-bit A 16x32: lanes 0-15 / 16-31 both carry M=0..15; VGPR j holds
    // K = (j<4 ? 2j : 16+2(j-4)) + 8*khalf  and K+1 (packed lo/hi 16 bits).
    v16bf ah[KSTEPS], al[KSTEPS];
#pragma unroll
    for (int ks = 0; ks < KSTEPS; ++ks) {
#pragma unroll
        for (int j = 0; j < 8; ++j) {
            const int kk = ((j < 4) ? (2 * j) : (16 + 2 * (j - 4))) + 8 * khalf;
#pragma unroll
            for (int h = 0; h < 2; ++h) {
                const int d = ks * 32 + kk + h;
                float f = zb[(size_t)d * HWSZ + m];
                __bf16 hb = (__bf16)f;
                ah[ks][2 * j + h] = hb;
                al[ks][2 * j + h] = (__bf16)(f - (float)hb);
            }
        }
    }

    // stage code tile nt into LDS buffer buf (16 KB = 1024 uint4)
    auto stage = [&](int nt, int buf) {
        const uint4* srcH = (const uint4*)(ehi + (size_t)nt * 4096);
        const uint4* srcL = (const uint4*)(elo + (size_t)nt * 4096);
        uint4* dst = lds4 + buf * 1024;
#if HAVE_ASYNC_LDS
#pragma unroll
        for (int i = 0; i < 2; ++i) {
            __builtin_amdgcn_global_load_async_to_lds_b128(
                (as1_v4i)(as1_ptr)(void*)(srcH + tid + i * 256),
                (as3_v4i)(as3_ptr)(void*)(dst + tid + i * 256), 0, 0);
            __builtin_amdgcn_global_load_async_to_lds_b128(
                (as1_v4i)(as1_ptr)(void*)(srcL + tid + i * 256),
                (as3_v4i)(as3_ptr)(void*)(dst + 512 + tid + i * 256), 0, 0);
        }
#else
        dst[tid]       = srcH[tid];
        dst[tid + 256] = srcH[tid + 256];
        dst[tid + 512] = srcL[tid];
        dst[tid + 768] = srcL[tid + 256];
#endif
    };

    unsigned long long best[8];
#pragma unroll
    for (int r = 0; r < 8; ++r) best[r] = ~0ULL;

    // prologue: stage tile 0 into buffer 0
    stage(0, 0);
    async_wait0();
    __syncthreads();

    for (int nt = 0; nt < NTILES; ++nt) {
        const int cur = nt & 1;
        // prefetch next tile into the other buffer (safe: all waves finished
        // reading it at iteration nt-1, enforced by the barrier below)
        if (nt + 1 < NTILES) stage(nt + 1, cur ^ 1);

        const __bf16* lh = (const __bf16*)(lds4 + cur * 1024);
        const __bf16* ll = (const __bf16*)(lds4 + cur * 1024 + 512);

        // 3 independent accumulator chains (hi*hi, hi*lo, lo*hi)
        v8f a0 = {}, a1 = {}, a2 = {};
#pragma unroll
        for (int ks = 0; ks < KSTEPS; ++ks) {
            v16bf bh = *(const v16bf*)(lh + (ks * 32 + lane) * 16);
            v16bf bl = *(const v16bf*)(ll + (ks * 32 + lane) * 16);
            a0 = __builtin_amdgcn_wmma_f32_16x16x32_bf16(false, ah[ks], false, bh,
                                                         (short)0, a0, false, false);
            a1 = __builtin_amdgcn_wmma_f32_16x16x32_bf16(false, ah[ks], false, bl,
                                                         (short)0, a1, false, false);
            a2 = __builtin_amdgcn_wmma_f32_16x16x32_bf16(false, al[ks], false, bh,
                                                         (short)0, a2, false, false);
        }

        // score(row, code) = e_sq[code] - 2 * dot  (z_sq is argmin-invariant)
        const unsigned int code = (unsigned int)(nt * 16 + m);   // column this lane holds
        const float es = esq[code];
#pragma unroll
        for (int r = 0; r < 8; ++r) {
            float dot = a0[r] + a1[r] + a2[r];
            float score = es - 2.0f * dot;
            unsigned long long key =
                ((unsigned long long)ord_f32(score) << 32) | (unsigned long long)code;
            if (key < best[r]) best[r] = key;
        }

        // prefetch completed + everyone done reading current buffer
        async_wait0();
        __syncthreads();
    }

    // cross-lane argmin: row M lives on 16 lanes (one column each).
    // C/D layout: VGPR r -> M=r (lanes 0-15), M=r+8 (lanes 16-31).
#pragma unroll
    for (int r = 0; r < 8; ++r) {
#pragma unroll
        for (int mask = 8; mask >= 1; mask >>= 1) {
            unsigned long long o = shfl_xor_u64(best[r], mask);
            if (o < best[r]) best[r] = o;
        }
    }

    const unsigned int n0 = (unsigned int)(b * HWSZ + hw0);
    if (lane == 0) {
#pragma unroll
        for (int r = 0; r < 8; ++r) idxout[n0 + r] = (unsigned int)best[r];
    } else if (lane == 16) {
#pragma unroll
        for (int r = 0; r < 8; ++r) idxout[n0 + 8 + r] = (unsigned int)best[r];
    }
}

// ---------------------------------------------------------------------------
// Kernel 3: gather z_q, write [B,C,H,W] output, per-block loss partials.
// 256 blocks x 256 threads; thread -> one row n (coalesced over hw per c).
// ---------------------------------------------------------------------------
__global__ void __launch_bounds__(256)
vq_gather(const float* __restrict__ z, const float* __restrict__ emb,
          const unsigned int* __restrict__ idx,
          float* __restrict__ out, float* __restrict__ partial) {
    __shared__ float sm[256];
    const int tid = threadIdx.x;
    const int n   = blockIdx.x * 256 + tid;      // 0..65535
    const int b   = n >> 10;
    const int hw  = n & 1023;
    const unsigned int code = idx[n];
    const float* e = emb + (size_t)code * LATENT;
    const size_t base = (size_t)b * (LATENT * HWSZ) + hw;

    float s = 0.0f;
#pragma unroll 4
    for (int c = 0; c < LATENT; ++c) {
        float zv = z[base + (size_t)c * HWSZ];
        float q  = e[c];
        out[base + (size_t)c * HWSZ] = q;
        float d = q - zv;
        s += d * d;
    }

    sm[tid] = s;
    __syncthreads();
    for (int r = 128; r > 0; r >>= 1) {
        if (tid < r) sm[tid] += sm[tid + r];
        __syncthreads();
    }
    if (tid == 0) partial[blockIdx.x] = sm[0];
}

// ---------------------------------------------------------------------------
// Kernel 4: final deterministic loss reduction.
// loss = (1 + BETA) * mean((z_q - z)^2), BETA = 0.25, mean over B*H*W*C.
// ---------------------------------------------------------------------------
__global__ void vq_loss(const float* __restrict__ partial, float* __restrict__ loss_out) {
    __shared__ float sm[256];
    const int tid = threadIdx.x;
    sm[tid] = partial[tid];
    __syncthreads();
    for (int r = 128; r > 0; r >>= 1) {
        if (tid < r) sm[tid] += sm[tid + r];
        __syncthreads();
    }
    if (tid == 0) loss_out[0] = 1.25f * sm[0] / 16777216.0f;
}

// ---------------------------------------------------------------------------
extern "C" void kernel_launch(void* const* d_in, const int* in_sizes, int n_in,
                              void* d_out, int out_size, void* d_ws, size_t ws_size,
                              hipStream_t stream) {
    const float* z   = (const float*)d_in[0];   // 16777216 f32
    const float* emb = (const float*)d_in[1];   // 262144 f32
    float* out = (float*)d_out;                 // 16777216 z_q + 1 loss

    char* ws = (char*)d_ws;
    __bf16*       ehi  = (__bf16*)(ws + WS_EHI);
    __bf16*       elo  = (__bf16*)(ws + WS_ELO);
    float*        esq  = (float*)(ws + WS_ESQ);
    unsigned int* idx  = (unsigned int*)(ws + WS_IDX);
    float*        part = (float*)(ws + WS_PART);

    vq_prep<<<N_EMB, 256, 0, stream>>>(emb, ehi, elo, esq);
    vq_dist<<<NROWS / 16 / WAVES_PER_BLOCK, 256, 0, stream>>>(z, ehi, elo, esq, idx);
    vq_gather<<<NROWS / 256, 256, 0, stream>>>(z, emb, idx, out, part);
    vq_loss<<<1, 256, 0, stream>>>(part, out + 16777216);
}
